// MultiHeadAttention_62732292325607
// MI455X (gfx1250) — compile-verified, matrix operand order checked by reading, not run on hardware
//
#include <hip/hip_runtime.h>
#include <hip/hip_bf16.h>
#include <stdint.h>

// ---------------- CDNA5 (gfx1250) types ----------------
typedef __bf16 bf16_t;
typedef __attribute__((ext_vector_type(16))) __bf16 v16bf;
typedef __attribute__((ext_vector_type(8)))  __bf16 v8bf;
typedef __attribute__((ext_vector_type(8)))  float  v8f;
typedef __attribute__((ext_vector_type(4)))  unsigned int u32x4;
typedef __attribute__((ext_vector_type(8)))  int i32x8;
typedef __attribute__((ext_vector_type(4)))  int i32x4;

#define HIDDEN 1024
#define NH     16
#define DK     64
#define BATCH  2
#define SEQ    2048
#define MTOT   (BATCH * SEQ)   // 4096 rows

#if defined(__gfx1250__) && __has_builtin(__builtin_amdgcn_tensor_load_to_lds) && \
    __has_builtin(__builtin_amdgcn_s_wait_tensorcnt)
#define USE_TDM 1
#else
#define USE_TDM 0
#endif

__device__ __forceinline__ v8f vzero8() {
  v8f z;
#pragma unroll
  for (int i = 0; i < 8; ++i) z[i] = 0.0f;
  return z;
}

__device__ __forceinline__ v16bf ones16() {
  v16bf f;
#pragma unroll
  for (int i = 0; i < 16; ++i) f[i] = (bf16_t)1.0f;
  return f;
}

// Load a 16-element bf16 fragment in the CDNA5 A/B register layout for
// 16x16x32: per lane, elements {kb..kb+7} and {kb+16..kb+23} of a row.
// p must be 16-byte aligned (tile strides are multiples of 8 elements).
__device__ __forceinline__ v16bf ld_frag(const bf16_t* p) {
  v8bf lo = *(const v8bf*)(p);
  v8bf hi = *(const v8bf*)(p + 16);
  v16bf f;
#pragma unroll
  for (int i = 0; i < 8; ++i) { f[i] = lo[i]; f[i + 8] = hi[i]; }
  return f;
}

__device__ __forceinline__ v8f wmma_bf16(v16bf a, v16bf b, v8f c) {
  // (neg_a, A, neg_b, B, c_mod, C, reuse_a, reuse_b)
  return __builtin_amdgcn_wmma_f32_16x16x32_bf16(false, a, false, b,
                                                 (short)0, c, false, false);
}

#if USE_TDM
// Issue a TDM 2D tile load (global -> LDS) per CDNA5 ISA.
// rows x cols bf16 tile, global row stride = stride_elems, with LDS row
// padding: after 2^(pad_interval_code+1) DWORDs insert pad_amount_code+1 DWORDs.
// Tracked by TENSORcnt.
__device__ __forceinline__ void tdm_load_2d(const void* gptr, void* lptr,
                                            unsigned rows, unsigned cols,
                                            unsigned long long stride_elems,
                                            unsigned pad_interval_code,
                                            unsigned pad_amount_code) {
  unsigned long long ga = (unsigned long long)(uintptr_t)gptr;
  unsigned int la = (unsigned int)(uintptr_t)lptr;  // LDS byte address
  u32x4 g0;
  g0[0] = 1u;                                   // count=1 (valid user D#)
  g0[1] = la;                                   // lds_addr        [63:32]
  g0[2] = (unsigned)(ga & 0xffffffffu);         // global_addr lo  [95:64]
  g0[3] = (unsigned)((ga >> 32) & 0x01ffffffu)  // global_addr hi  [120:96]
          | 0x80000000u;                        // type=2 ("image")[127:126]
  i32x8 g1;
  g1[0] = (int)((1u << 16)                      // data_size = 2 bytes
                | (1u << 20)                    // pad_enable
                | ((pad_interval_code & 7u) << 22)
                | ((pad_amount_code & 127u) << 25));
  unsigned long long td0 = stride_elems;        // tensor_dim0 (>= tile cols)
  unsigned long long td1 = (unsigned long long)rows;
  g1[1] = (int)((td0 & 0xffffu) << 16);                               // dim0 lo16
  g1[2] = (int)(((td0 >> 16) & 0xffffu) | ((td1 & 0xffffu) << 16));   // dim0 hi | dim1 lo
  g1[3] = (int)(((td1 >> 16) & 0xffffu) | ((cols & 0xffffu) << 16));  // dim1 hi | tile_dim0
  g1[4] = (int)(rows & 0xffffu);                                      // tile_dim1 (tile_dim2=0)
  g1[5] = (int)(stride_elems & 0xffffffffu);                          // dim0_stride lo32
  g1[6] = (int)((stride_elems >> 32) & 0xffffu);                      // dim0_stride hi16
  g1[7] = 0;
  i32x4 z4 = {0, 0, 0, 0};
#if __clang_major__ >= 23
  i32x8 z8 = {0, 0, 0, 0, 0, 0, 0, 0};
  __builtin_amdgcn_tensor_load_to_lds(g0, g1, z4, z4, z8, 0);
#else
  __builtin_amdgcn_tensor_load_to_lds(g0, g1, z4, z4, 0);
#endif
}
#endif  // USE_TDM

// ---------------- fp32 -> bf16 conversion ----------------
__global__ void mha_cvt_bf16(const float* __restrict__ in,
                             bf16_t* __restrict__ out, size_t n) {
  size_t i = (size_t)blockIdx.x * blockDim.x + threadIdx.x;
  size_t stride = (size_t)gridDim.x * blockDim.x;
  for (; i < n; i += stride) out[i] = (bf16_t)in[i];
}

// ---------------- fp32 [K][N] -> bf16 transposed [N][K] ----------------
__global__ __launch_bounds__(256) void mha_cvt_transpose(
    const float* __restrict__ in, bf16_t* __restrict__ out, int K, int N) {
  __shared__ float tile[32][33];
  int kb = blockIdx.y * 32, nb = blockIdx.x * 32;
  int tx = threadIdx.x & 31, ty = threadIdx.x >> 5;  // 8 rows per pass
#pragma unroll
  for (int r = ty; r < 32; r += 8)
    tile[r][tx] = in[(size_t)(kb + r) * N + (nb + tx)];
  __syncthreads();
#pragma unroll
  for (int r = ty; r < 32; r += 8)
    out[(size_t)(nb + r) * K + (kb + tx)] = (bf16_t)tile[tx][r];
}

// ---------------- tiled bf16 GEMM, f32 accumulate -----------------
// C[M x NDIM] = A[M x 1024] * Bt[NDIM x 1024]^T + bias  (Bt pre-transposed)
// Block: 256 threads (8 waves), tile 128(M) x 128(N), K-step 32,
// double-buffered TDM staging (DMA of tile i+1 overlaps compute of tile i).
// TO_QKV: scatter bf16 into q/k [sel][b][h][t][d], v as [2][b][h][d][t].
template <int NDIM, bool TO_QKV>
__global__ __launch_bounds__(256) void mha_gemm_bf16(
    const bf16_t* __restrict__ A, const bf16_t* __restrict__ Bt,
    const float* __restrict__ bias, float* __restrict__ outF,
    bf16_t* __restrict__ outQ) {
  const int K = HIDDEN;
  __shared__ __align__(16) bf16_t As[2][128][40];   // [buf][m][k], stride 80B
  __shared__ __align__(16) bf16_t Bs[2][128][40];   // [buf][n][k]

  const int bm = blockIdx.y * 128, bn = blockIdx.x * 128;
  const int tid = threadIdx.x, lane = tid & 31, w = tid >> 5;
  const int nloc = lane & 15, half = lane >> 4, kb = half * 8;

  v8f acc[8];
#pragma unroll
  for (int c = 0; c < 8; ++c) acc[c] = vzero8();

  int cur = 0;
#if USE_TDM
  if (w == 0) {  // one wave drives the Tensor Data Mover
    tdm_load_2d(&A[(size_t)bm * K], &As[0][0][0], 128, 32, K, 3, 3);
    tdm_load_2d(&Bt[(size_t)bn * K], &Bs[0][0][0], 128, 32, K, 3, 3);
    __builtin_amdgcn_s_wait_tensorcnt(0);
  }
  __syncthreads();
#endif

  for (int k0 = 0; k0 < K; k0 += 32) {
#if USE_TDM
    if (w == 0 && k0 + 32 < K) {  // prefetch next tile into alternate buffer
      tdm_load_2d(&A[(size_t)bm * K + k0 + 32], &As[cur ^ 1][0][0], 128, 32, K, 3, 3);
      tdm_load_2d(&Bt[(size_t)bn * K + k0 + 32], &Bs[cur ^ 1][0][0], 128, 32, K, 3, 3);
    }
#else
#pragma unroll
    for (int it = 0; it < 16; ++it) {
      int i = it * 256 + tid;
      int m = i >> 5, kk = i & 31;
      As[0][m][kk] = A[(size_t)(bm + m) * K + (k0 + kk)];
      Bs[0][m][kk] = Bt[(size_t)(bn + m) * K + (k0 + kk)];
    }
    __syncthreads();
#endif

    v16bf af = ld_frag(&As[cur][w * 16 + nloc][kb]);
#pragma unroll
    for (int c = 0; c < 8; ++c) {
      v16bf bf = ld_frag(&Bs[cur][c * 16 + nloc][kb]);
      acc[c] = wmma_bf16(af, bf, acc[c]);
    }

#if USE_TDM
    if (w == 0 && k0 + 32 < K) __builtin_amdgcn_s_wait_tensorcnt(0);
    __syncthreads();
    cur ^= 1;
#else
    __syncthreads();
#endif
  }

  // epilogue; C layout: row = r + 8*half, col = c*16 + nloc
#pragma unroll
  for (int c = 0; c < 8; ++c) {
    int col = bn + c * 16 + nloc;
#pragma unroll
    for (int r = 0; r < 8; ++r) {
      int mrow = bm + w * 16 + r + half * 8;
      float v = acc[c][r] + bias[col];
      if (TO_QKV) {
        int sel = col >> 10;                  // 0=q 1=k 2=v
        int cc = col & 1023;
        int h = cc >> 6, d = cc & 63;
        int b = mrow >> 11, t = mrow & 2047;  // SEQ = 2048
        size_t base = (size_t)sel * BATCH * NH * SEQ * DK;
        size_t inner = (sel == 2)
            ? (((size_t)(b * NH + h) * DK + d) * SEQ + t)   // V: d-major
            : (((size_t)(b * NH + h) * SEQ + t) * DK + d);  // Q,K: t-major
        outQ[base + inner] = (bf16_t)v;
      } else {
        outF[(size_t)mrow * NDIM + col] = v;
      }
    }
  }
}

// ---------------- flash attention ----------------
// grid: (SEQ/128, NH, BATCH); 256 threads = 8 waves, 16 query rows per wave.
// q,k: [b][h][t][d] bf16; v: [b][h][d][t] bf16 (pre-transposed by GEMM1).
// K/V tiles double-buffered via TDM; P row-sum done on the matrix pipe.
__global__ __launch_bounds__(256) void mha_attn(
    const bf16_t* __restrict__ q, const bf16_t* __restrict__ k,
    const bf16_t* __restrict__ v, const int* __restrict__ mask,
    bf16_t* __restrict__ attn_out) {
  __shared__ __align__(16) bf16_t qs[128][72];     // Q tile, row-major
  __shared__ __align__(16) bf16_t kt[2][32][72];   // K tile, [buf][key][dk]
  __shared__ __align__(16) bf16_t vt[2][64][40];   // V tile, [buf][d][key]
  __shared__ __align__(16) bf16_t ps[8][16][40];   // per-wave P scratch [m][key]

  const int qb = blockIdx.x * 128;
  const int h = blockIdx.y, b = blockIdx.z;
  const int tid = threadIdx.x, lane = tid & 31, w = tid >> 5;
  const int nloc = lane & 15, half = lane >> 4, kb = half * 8;

  const bf16_t* qh = q + ((size_t)(b * NH + h)) * SEQ * DK;
  const bf16_t* kh = k + ((size_t)(b * NH + h)) * SEQ * DK;
  const bf16_t* vh = v + ((size_t)(b * NH + h)) * SEQ * DK;  // [d][t]
  const int* mrow_base = mask + (size_t)b * SEQ;

  int cur = 0;
#if USE_TDM
  if (w == 0) {
    tdm_load_2d(&qh[(size_t)qb * DK], &qs[0][0], 128, 64, DK, 4, 3);
    tdm_load_2d(&kh[0], &kt[0][0][0], 32, 64, DK, 4, 3);
    tdm_load_2d(&vh[0], &vt[0][0][0], 64, 32, SEQ, 3, 3);
    __builtin_amdgcn_s_wait_tensorcnt(0);
  }
#else
#pragma unroll
  for (int it = 0; it < 32; ++it) {
    int i = it * 256 + tid;
    int m = i >> 6, d = i & 63;
    qs[m][d] = qh[(size_t)(qb + m) * DK + d];
  }
#pragma unroll
  for (int it = 0; it < 8; ++it) {
    int i = it * 256 + tid;
    int kk = i >> 6, d = i & 63;
    kt[0][kk][d] = kh[(size_t)kk * DK + d];
    int dd = i >> 5, tt = i & 31;
    vt[0][dd][tt] = vh[(size_t)dd * SEQ + tt];
  }
#endif
  __syncthreads();

  // persistent Q fragments: dk 0..31 and 32..63
  v16bf qf0 = ld_frag(&qs[w * 16 + nloc][kb]);
  v16bf qf1 = ld_frag(&qs[w * 16 + nloc][32 + kb]);
  const v16bf onef = ones16();

  float mx[8], lsum[8];
#pragma unroll
  for (int r = 0; r < 8; ++r) { mx[r] = -1e30f; lsum[r] = 0.0f; }
  v8f o[4];
#pragma unroll
  for (int c = 0; c < 4; ++c) o[c] = vzero8();

  for (int kb0 = 0; kb0 < SEQ; kb0 += 32) {
#if USE_TDM
    if (w == 0 && kb0 + 32 < SEQ) {  // prefetch next K/V tile
      tdm_load_2d(&kh[(size_t)(kb0 + 32) * DK], &kt[cur ^ 1][0][0], 32, 64, DK, 4, 3);
      tdm_load_2d(&vh[kb0 + 32], &vt[cur ^ 1][0][0], 64, 32, SEQ, 3, 3);
    }
#else
    if (kb0 > 0) {
#pragma unroll
      for (int it = 0; it < 8; ++it) {
        int i = it * 256 + tid;
        int kk = i >> 6, d = i & 63;
        kt[0][kk][d] = kh[(size_t)(kb0 + kk) * DK + d];
        int dd = i >> 5, tt = i & 31;
        vt[0][dd][tt] = vh[(size_t)dd * SEQ + kb0 + tt];
      }
    }
    __syncthreads();
#endif

    // S = Q * K^T : two 16x16 C tiles (keys 0-15, 16-31)
    v8f s0 = vzero8(), s1 = vzero8();
    {
      v16bf k00 = ld_frag(&kt[cur][nloc][kb]);
      v16bf k01 = ld_frag(&kt[cur][nloc][32 + kb]);
      s0 = wmma_bf16(qf0, k00, s0);
      s0 = wmma_bf16(qf1, k01, s0);
      v16bf k10 = ld_frag(&kt[cur][16 + nloc][kb]);
      v16bf k11 = ld_frag(&kt[cur][16 + nloc][32 + kb]);
      s1 = wmma_bf16(qf0, k10, s1);
      s1 = wmma_bf16(qf1, k11, s1);
    }

    // scale + mask (mask depends only on key column = lane)
    float mb0 = mrow_base[kb0 + nloc] ? 0.0f : -1e9f;
    float mb1 = mrow_base[kb0 + 16 + nloc] ? 0.0f : -1e9f;
#pragma unroll
    for (int r = 0; r < 8; ++r) {
      s0[r] = s0[r] * 0.125f + mb0;
      s1[r] = s1[r] * 0.125f + mb1;
    }

    // online softmax: row max via cross-lane reduction over each 16-lane half
    float alpha[8];
#pragma unroll
    for (int r = 0; r < 8; ++r) {
      float t = fmaxf(s0[r], s1[r]);
      t = fmaxf(t, __shfl_xor(t, 1, 32));
      t = fmaxf(t, __shfl_xor(t, 2, 32));
      t = fmaxf(t, __shfl_xor(t, 4, 32));
      t = fmaxf(t, __shfl_xor(t, 8, 32));
      float mnew = fmaxf(mx[r], t);
      alpha[r] = __expf(mx[r] - mnew);
      float p0 = __expf(s0[r] - mnew);
      float p1 = __expf(s1[r] - mnew);
      ps[w][r + half * 8][nloc] = (bf16_t)p0;        // C layout -> A layout
      ps[w][r + half * 8][16 + nloc] = (bf16_t)p1;   // via per-wave LDS
      mx[r] = mnew;
    }

    // rescale running output
#pragma unroll
    for (int c = 0; c < 4; ++c)
#pragma unroll
      for (int r = 0; r < 8; ++r) o[c][r] *= alpha[r];

    // P fragment (A layout) feeds both the row-sum and the PV matmul
    v16bf pf = ld_frag(&ps[w][nloc][kb]);

    // row-sum of P on the matrix pipe: P(16x32) x ones(32x16) -> every
    // element of row m holds sum_k P[m][k]; each lane reads its own [r].
    v8f rowsum = wmma_bf16(pf, onef, vzero8());
#pragma unroll
    for (int r = 0; r < 8; ++r) lsum[r] = lsum[r] * alpha[r] + rowsum[r];

    // O += P(16x32) * V(32x64)
#pragma unroll
    for (int c = 0; c < 4; ++c) {
      v16bf vf = ld_frag(&vt[cur][c * 16 + nloc][kb]);
      o[c] = wmma_bf16(pf, vf, o[c]);
    }

#if USE_TDM
    if (w == 0 && kb0 + 32 < SEQ) __builtin_amdgcn_s_wait_tensorcnt(0);
    __syncthreads();
    cur ^= 1;
#else
    __syncthreads();
#endif
  }

  // normalize and store bf16 into [B*T][HIDDEN] for the output projection
  float inv[8];
#pragma unroll
  for (int r = 0; r < 8; ++r) inv[r] = 1.0f / lsum[r];
#pragma unroll
  for (int c = 0; c < 4; ++c) {
    int col = h * DK + c * 16 + nloc;
#pragma unroll
    for (int r = 0; r < 8; ++r) {
      int t = qb + w * 16 + r + half * 8;
      attn_out[((size_t)(b * SEQ + t)) * HIDDEN + col] = (bf16_t)(o[c][r] * inv[r]);
    }
  }
}

// ---------------- launcher ----------------
extern "C" void kernel_launch(void* const* d_in, const int* in_sizes, int n_in,
                              void* d_out, int out_size, void* d_ws, size_t ws_size,
                              hipStream_t stream) {
  const float* x    = (const float*)d_in[0];
  const int*   mask = (const int*)d_in[1];
  const float* Wqkv = (const float*)d_in[2];
  const float* bqkv = (const float*)d_in[3];
  const float* Wout = (const float*)d_in[4];
  const float* bout = (const float*)d_in[5];
  float* out = (float*)d_out;

  bf16_t* ws = (bf16_t*)d_ws;
  size_t o = 0;
  bf16_t* xb    = ws + o; o += (size_t)MTOT * HIDDEN;             // 4M
  bf16_t* wqkvT = ws + o; o += (size_t)HIDDEN * 3 * HIDDEN;       // 3M  [3072][1024]
  bf16_t* woutT = ws + o; o += (size_t)HIDDEN * HIDDEN;           // 1M  [1024][1024]
  bf16_t* qkvb  = ws + o; o += (size_t)3 * BATCH * NH * SEQ * DK; // 12M
  bf16_t* attnb = ws + o; o += (size_t)MTOT * HIDDEN;             // 4M  (~48MB bf16)

  mha_cvt_bf16<<<2048, 256, 0, stream>>>(x, xb, (size_t)MTOT * HIDDEN);
  mha_cvt_transpose<<<dim3(3 * HIDDEN / 32, HIDDEN / 32), 256, 0, stream>>>(
      Wqkv, wqkvT, HIDDEN, 3 * HIDDEN);
  mha_cvt_transpose<<<dim3(HIDDEN / 32, HIDDEN / 32), 256, 0, stream>>>(
      Wout, woutT, HIDDEN, HIDDEN);

  // qkv = x @ W_qkv + b_qkv  -> bf16, q/k t-major, v d-major
  mha_gemm_bf16<3 * HIDDEN, true>
      <<<dim3(3 * HIDDEN / 128, MTOT / 128), 256, 0, stream>>>(
          xb, wqkvT, bqkv, nullptr, qkvb);

  size_t selStride = (size_t)BATCH * NH * SEQ * DK;
  mha_attn<<<dim3(SEQ / 128, NH, BATCH), 256, 0, stream>>>(
      qkvb, qkvb + selStride, qkvb + 2 * selStride, mask, attnb);

  // out = attn @ W_out + b_out -> fp32
  mha_gemm_bf16<HIDDEN, false>
      <<<dim3(HIDDEN / 128, MTOT / 128), 256, 0, stream>>>(
          attnb, woutT, bout, out, nullptr);
}
